// CognitiveNetwork_23716809409310
// MI455X (gfx1250) — matrix operand order repaired and checked by reading, not compiled, not run on hardware
//
#include <hip/hip_runtime.h>
#include <hip/hip_bf16.h>

// ---------------------------------------------------------------------------
// CognitiveNetwork on MI455X (gfx1250): bf16 WMMA pipeline, software-pipelined.
// Weights converted once to bf16 [N,K]; per-timestep kernels use
// v_wmma_f32_16x16x32_bf16 with fp32 accumulation; LSTM state c in fp32,
// h carried in bf16 (double-buffered). Kernel launch boundaries are the
// global barriers the cross-cell recurrence (ext = mean over cells) needs.
// ---------------------------------------------------------------------------

typedef __bf16 bf16_t;
typedef __attribute__((ext_vector_type(16))) __bf16 v16bf;
typedef __attribute__((ext_vector_type(8)))  __bf16 v8bf;
typedef __attribute__((ext_vector_type(8)))  float  v8f;

static constexpr int Ee = 256;
static constexpr int Hh = 512;
static constexpr int Cc = 16;
static constexpr int Bb = 256;
static constexpr int Tt = 128;
static constexpr int G4H = 4 * Hh;   // 2048

__device__ __forceinline__ float sigmoid_f(float x) {
    return 1.0f / (1.0f + expf(-x));
}

// ---- WMMA fragment loaders (bf16, 16x16x32) -------------------------------
// A matrix 16x32 (MxK), row-major source, stride lda (halfs):
//   lane L holds row m=L&15; halves 0..7 -> K=(L>>4)*8+0..7, halves 8..15 -> +16
__device__ __forceinline__ v16bf load_frag_a(const bf16_t* base, int lda, int lane) {
    int m  = lane & 15;
    int ko = (lane >> 4) << 3;
    const v8bf* p0 = (const v8bf*)(base + (size_t)m * lda + ko);
    const v8bf* p1 = (const v8bf*)(base + (size_t)m * lda + ko + 16);
    v8bf a0 = *p0, a1 = *p1;
    v16bf f;
#pragma unroll
    for (int i = 0; i < 8; ++i) { f[i] = a0[i]; f[i + 8] = a1[i]; }
    return f;
}

// B matrix 32x16 (KxN), weights stored transposed as [N,K] row-major:
//   lane L holds col n=L&15; halves 0..7 -> K=(L>>4)*8+0..7, halves 8..15 -> +16
__device__ __forceinline__ v16bf load_frag_b(const bf16_t* baseNK, int ldb, int lane) {
    int n  = lane & 15;
    int ko = (lane >> 4) << 3;
    const v8bf* p0 = (const v8bf*)(baseNK + (size_t)n * ldb + ko);
    v8bf a0 = p0[0], a1 = p0[2];
    v16bf f;
#pragma unroll
    for (int i = 0; i < 8; ++i) { f[i] = a0[i]; f[i + 8] = a1[i]; }
    return f;
}

#define WMMA_BF16(a, b, c) \
    __builtin_amdgcn_wmma_f32_16x16x32_bf16(false, (a), false, (b), (short)0, (c), false, false)

// ---------------------------------------------------------------------------
// Prep kernels
// ---------------------------------------------------------------------------
__global__ void k_zero_f32(float* p, int n) {
    int i = blockIdx.x * 256 + threadIdx.x;
    if (i < n) p[i] = 0.0f;
}
__global__ void k_zero_bf(bf16_t* p, int n) {
    int i = blockIdx.x * 256 + threadIdx.x;
    if (i < n) p[i] = (bf16_t)0.0f;
}
__global__ void k_cvt_bf16(const float* __restrict__ s, bf16_t* __restrict__ d, int n) {
    int i = blockIdx.x * 256 + threadIdx.x;
    if (i < n) d[i] = (bf16_t)s[i];
}
// dst[c,n,k] = src[c,k,n]   (src: [C,H,H] with out = x @ W  ->  B[N,K] = W[K,N])
__global__ void k_cvt_transpose_chh(const float* __restrict__ s, bf16_t* __restrict__ d, int n) {
    int i = blockIdx.x * 256 + threadIdx.x;
    if (i < n) {
        int c = i / (Hh * Hh), r = i % (Hh * Hh);
        int nn = r / Hh, kk = r % Hh;
        d[i] = (bf16_t)s[(size_t)c * Hh * Hh + (size_t)kk * Hh + nn];
    }
}
// dst[n,e] = src[e,n]   (Wproj [E,H] -> [H,E])
__global__ void k_cvt_transpose_proj(const float* __restrict__ s, bf16_t* __restrict__ d, int n) {
    int i = blockIdx.x * 256 + threadIdx.x;
    if (i < n) {
        int nn = i / Ee, e = i % Ee;
        d[i] = (bf16_t)s[(size_t)e * Hh + nn];
    }
}
__global__ void k_bias_sum(const float* __restrict__ a, const float* __restrict__ b,
                           float* __restrict__ o, int n) {
    int i = blockIdx.x * 256 + threadIdx.x;
    if (i < n) o[i] = a[i] + b[i];
}

// ---------------------------------------------------------------------------
// Embedding gather + projection: xs[t,b,:] = emb[tok(b,t)] @ Wproj + bproj
// M = T*B = 32768 rows, K = E = 256, N = H = 512. One block = 16 rows (mtile),
// 8 waves x 4 ntiles cover N. emb rows converted fp32->bf16 on the fly.
// A and B fragments double-buffered so gather/convert overlaps the WMMAs.
// ---------------------------------------------------------------------------
__device__ __forceinline__ v16bf build_emb_frag(const float* erow, int tok, int kb) {
    v16bf a = {};
    if (tok != 0) {                           // padding_idx=0 -> zero row
#pragma unroll
        for (int i = 0; i < 8; ++i) {
            a[i]     = (bf16_t)erow[kb + i];
            a[i + 8] = (bf16_t)erow[kb + 16 + i];
        }
    }
    return a;
}

__global__ __launch_bounds__(256) void k_embed_proj(
    const int* __restrict__ tokens, const float* __restrict__ emb,
    const bf16_t* __restrict__ WprojT, const float* __restrict__ bproj,
    float* __restrict__ xs) {
    int mtile = blockIdx.x;
    int tid = threadIdx.x, lane = tid & 31, wave = tid >> 5;
    int m  = lane & 15;
    int rg = mtile * 16 + m;                 // row = t*B + b
    int tt = rg >> 8, bb = rg & 255;
    int tok = tokens[bb * Tt + tt];
    const float* erow = emb + (size_t)tok * Ee;
    int kol = (lane >> 4) << 3;

    v8f acc[4] = {};
    v16bf acur = build_emb_frag(erow, tok, kol);
    v16bf bcur[4];
#pragma unroll
    for (int i = 0; i < 4; ++i)
        bcur[i] = load_frag_b(WprojT + (size_t)((wave * 4 + i) * 16) * Ee, Ee, lane);

    for (int k = 0; k < 8; ++k) {
        v16bf anxt;
        v16bf bnxt[4];
        if (k < 7) {
            anxt = build_emb_frag(erow, tok, (k + 1) * 32 + kol);
#pragma unroll
            for (int i = 0; i < 4; ++i)
                bnxt[i] = load_frag_b(WprojT + (size_t)((wave * 4 + i) * 16) * Ee + (k + 1) * 32,
                                      Ee, lane);
        }
#pragma unroll
        for (int i = 0; i < 4; ++i)
            acc[i] = WMMA_BF16(acur, bcur[i], acc[i]);
        if (k < 7) {
            acur = anxt;
#pragma unroll
            for (int i = 0; i < 4; ++i) bcur[i] = bnxt[i];
        }
    }
#pragma unroll
    for (int i = 0; i < 4; ++i) {
        int j = (wave * 4 + i) * 16 + (lane & 15);
        float bj = bproj[j];
#pragma unroll
        for (int r = 0; r < 8; ++r) {
            int m2 = ((lane >> 4) << 3) + r;
            int rg2 = mtile * 16 + m2;
            int t2 = rg2 >> 8, b2 = rg2 & 255;
            xs[((size_t)t2 * Bb + b2) * Hh + j] = acc[i][r] + bj;
        }
    }
}

// ---------------------------------------------------------------------------
// k1: perception. Block = (cell, mtile of 16 batch rows).
//   X = x_t + 0.3*ext (shared by all cells) staged to LDS as bf16 (b128 ds),
//   P = LayerNorm(ReLU(X @ WpT + bp)) * ln_g + ln_b, stored bf16.
// B-fragments double-buffered so loads overlap the WMMAs.
// ---------------------------------------------------------------------------
__global__ __launch_bounds__(256) void k_perception(
    const float* __restrict__ xs_t,        // xs + t*B*H
    const float* __restrict__ ext, int ext_stride,
    const bf16_t* __restrict__ WpT,        // [C,H,H] (N,K)
    const float* __restrict__ bp,          // [C,H]
    const float* __restrict__ ln_g, const float* __restrict__ ln_b,
    bf16_t* __restrict__ Pbf) {            // [C,B,H]
    __shared__ __align__(16) bf16_t ldsX[16 * Hh];
    __shared__ float  ldsP[16 * Hh];
    __shared__ float  red[2 * 256];
    __shared__ float  mu_s[16], rs_s[16];

    int c = blockIdx.x >> 4, mtile = blockIdx.x & 15;
    int tid = threadIdx.x, lane = tid & 31, wave = tid >> 5;

    // stage X tile (16 x 512) into LDS as bf16: 32 contiguous cols per thread
    {
        int row = tid >> 4, cb = (tid & 15) * 32;
        int b = mtile * 16 + row;
        const float* xr = xs_t + (size_t)b * Hh + cb;
        const float* er = ext + (size_t)b * ext_stride + cb;
#pragma unroll
        for (int g = 0; g < 4; ++g) {
            v8bf pack;
#pragma unroll
            for (int i = 0; i < 8; ++i)
                pack[i] = (bf16_t)(xr[g * 8 + i] + 0.3f * er[g * 8 + i]);
            *(v8bf*)(ldsX + row * Hh + cb + g * 8) = pack;
        }
    }
    __syncthreads();

    const bf16_t* W = WpT + (size_t)c * Hh * Hh;
    v8f acc[4] = {};
    v16bf bcur[4];
#pragma unroll
    for (int i = 0; i < 4; ++i)
        bcur[i] = load_frag_b(W + (size_t)((wave * 4 + i) * 16) * Hh, Hh, lane);
    for (int k = 0; k < 16; ++k) {
        v16bf a = load_frag_a(ldsX + k * 32, Hh, lane);   // ds_load path
        v16bf bnxt[4];
        if (k < 15) {
#pragma unroll
            for (int i = 0; i < 4; ++i)
                bnxt[i] = load_frag_b(W + (size_t)((wave * 4 + i) * 16) * Hh + (k + 1) * 32, Hh, lane);
        }
#pragma unroll
        for (int i = 0; i < 4; ++i)
            acc[i] = WMMA_BF16(a, bcur[i], acc[i]);
        if (k < 15) {
#pragma unroll
            for (int i = 0; i < 4; ++i) bcur[i] = bnxt[i];
        }
    }
    // bias + ReLU -> LDS (fp32) for LayerNorm
#pragma unroll
    for (int i = 0; i < 4; ++i) {
        int j = (wave * 4 + i) * 16 + (lane & 15);
        float bj = bp[c * Hh + j];
#pragma unroll
        for (int r = 0; r < 8; ++r) {
            int m = ((lane >> 4) << 3) + r;
            ldsP[m * Hh + j] = fmaxf(acc[i][r] + bj, 0.0f);
        }
    }
    __syncthreads();

    // LayerNorm: 16 threads per row
    int row = tid >> 4, sub = tid & 15;
    float s = 0.0f, s2 = 0.0f;
#pragma unroll 4
    for (int q = 0; q < 32; ++q) {
        float v = ldsP[row * Hh + sub + q * 16];
        s += v; s2 += v * v;
    }
    red[row * 16 + sub] = s;
    red[256 + row * 16 + sub] = s2;
    __syncthreads();
    if (sub == 0) {
        float S = 0.0f, S2 = 0.0f;
#pragma unroll
        for (int q = 0; q < 16; ++q) { S += red[row * 16 + q]; S2 += red[256 + row * 16 + q]; }
        float mu = S * (1.0f / Hh);
        float var = S2 * (1.0f / Hh) - mu * mu;
        mu_s[row] = mu;
        rs_s[row] = rsqrtf(var + 1e-5f);
    }
    __syncthreads();

    // normalize + write bf16 P (b128 stores: 32 contiguous cols per thread)
    {
        int rr = tid >> 4, cb = (tid & 15) * 32;
        float mu = mu_s[rr], rs = rs_s[rr];
        bf16_t* dst = Pbf + ((size_t)c * Bb + mtile * 16 + rr) * Hh + cb;
#pragma unroll
        for (int g = 0; g < 4; ++g) {
            v8bf pack;
#pragma unroll
            for (int i = 0; i < 8; ++i) {
                int col = cb + g * 8 + i;
                float v = (ldsP[rr * Hh + col] - mu) * rs * ln_g[c * Hh + col] + ln_b[c * Hh + col];
                pack[i] = (bf16_t)v;
            }
            *(v8bf*)(dst + g * 8) = pack;
        }
    }
}

// ---------------------------------------------------------------------------
// k2: gates + fused LSTM (dominant kernel).
// Block = (cell, mgroup of 32 batch rows, jgroup). A-tiles (P then h, 32x512
// bf16 = 32KB) staged to LDS and shared by all 8 waves; each wave owns one
// 16-column slice j and computes the i/f/g/o tiles for BOTH 16-row halves
// (8 WMMAs per 4 B-fragment loads). B-fragments double-buffered.
// grid = C * 8 * 4 = 512 blocks.
// ---------------------------------------------------------------------------
__global__ __launch_bounds__(256) void k_gates_lstm(
    const bf16_t* __restrict__ Pbf,   // [C,B,H]
    const bf16_t* __restrict__ hprev, // [C,B,H]
    const bf16_t* __restrict__ Wihb,  // [C,4H,H] (N,K)
    const bf16_t* __restrict__ Whhb,  // [C,4H,H] (N,K)
    const float*  __restrict__ bihh,  // [C,4H] = bih + bhh
    float*  __restrict__ cstate,      // [C,B,H]
    bf16_t* __restrict__ hnew) {      // [C,B,H]
    __shared__ __align__(16) bf16_t ldsA[32 * Hh];   // 32KB, reused for P then h

    int bx = blockIdx.x;
    int c = bx >> 5, rem = bx & 31, mg = rem >> 2, jg = rem & 3;
    int m0 = mg * 32;
    int tid = threadIdx.x, lane = tid & 31, wave = tid >> 5;
    int jtile = jg * 8 + wave, j0 = jtile * 16;

    v8f acc[8] = {};   // [gate q][mtile half]: acc[q] = half0, acc[4+q] = half1

    const bf16_t* Wi = Wihb + (size_t)c * G4H * Hh;
    const bf16_t* Wh = Whhb + (size_t)c * G4H * Hh;

    // ---- pass 1: P @ Wih^T ----
    {
        const v8bf* src = (const v8bf*)(Pbf + ((size_t)c * Bb + m0) * Hh);
        v8bf* dst = (v8bf*)ldsA;
        for (int i = tid; i < (32 * Hh) / 8; i += 256) dst[i] = src[i];
    }
    __syncthreads();
    {
        v16bf bcur[4];
#pragma unroll
        for (int q = 0; q < 4; ++q)
            bcur[q] = load_frag_b(Wi + (size_t)(q * Hh + j0) * Hh, Hh, lane);
        for (int k = 0; k < 16; ++k) {
            v16bf a0 = load_frag_a(ldsA + k * 32, Hh, lane);
            v16bf a1 = load_frag_a(ldsA + 16 * Hh + k * 32, Hh, lane);
            v16bf bnxt[4];
            if (k < 15) {
#pragma unroll
                for (int q = 0; q < 4; ++q)
                    bnxt[q] = load_frag_b(Wi + (size_t)(q * Hh + j0) * Hh + (k + 1) * 32, Hh, lane);
            }
#pragma unroll
            for (int q = 0; q < 4; ++q) {
                acc[q]     = WMMA_BF16(a0, bcur[q], acc[q]);
                acc[4 + q] = WMMA_BF16(a1, bcur[q], acc[4 + q]);
            }
            if (k < 15) {
#pragma unroll
                for (int q = 0; q < 4; ++q) bcur[q] = bnxt[q];
            }
        }
    }
    __syncthreads();   // all waves done reading P tile

    // ---- pass 2: + h_prev @ Whh^T ----
    {
        const v8bf* src = (const v8bf*)(hprev + ((size_t)c * Bb + m0) * Hh);
        v8bf* dst = (v8bf*)ldsA;
        for (int i = tid; i < (32 * Hh) / 8; i += 256) dst[i] = src[i];
    }
    __syncthreads();
    {
        v16bf bcur[4];
#pragma unroll
        for (int q = 0; q < 4; ++q)
            bcur[q] = load_frag_b(Wh + (size_t)(q * Hh + j0) * Hh, Hh, lane);
        for (int k = 0; k < 16; ++k) {
            v16bf a0 = load_frag_a(ldsA + k * 32, Hh, lane);
            v16bf a1 = load_frag_a(ldsA + 16 * Hh + k * 32, Hh, lane);
            v16bf bnxt[4];
            if (k < 15) {
#pragma unroll
                for (int q = 0; q < 4; ++q)
                    bnxt[q] = load_frag_b(Wh + (size_t)(q * Hh + j0) * Hh + (k + 1) * 32, Hh, lane);
            }
#pragma unroll
            for (int q = 0; q < 4; ++q) {
                acc[q]     = WMMA_BF16(a0, bcur[q], acc[q]);
                acc[4 + q] = WMMA_BF16(a1, bcur[q], acc[4 + q]);
            }
            if (k < 15) {
#pragma unroll
                for (int q = 0; q < 4; ++q) bcur[q] = bnxt[q];
            }
        }
    }

    // ---- fused LSTM elementwise (torch gate order i, f, g, o) ----
    int n = lane & 15, j = j0 + n;
    float bi = bihh[c * G4H + 0 * Hh + j];
    float bf = bihh[c * G4H + 1 * Hh + j];
    float bg = bihh[c * G4H + 2 * Hh + j];
    float bo = bihh[c * G4H + 3 * Hh + j];
#pragma unroll
    for (int h = 0; h < 2; ++h) {
#pragma unroll
        for (int r = 0; r < 8; ++r) {
            int b = m0 + h * 16 + ((lane >> 4) << 3) + r;
            size_t idx = ((size_t)c * Bb + b) * Hh + j;
            float ig = sigmoid_f(acc[4 * h + 0][r] + bi);
            float fg = sigmoid_f(acc[4 * h + 1][r] + bf);
            float gg = tanhf(acc[4 * h + 2][r] + bg);
            float og = sigmoid_f(acc[4 * h + 3][r] + bo);
            float cn = fg * cstate[idx] + ig * gg;
            cstate[idx] = cn;
            hnew[idx] = (bf16_t)(og * tanhf(cn));
        }
    }
}

// ---------------------------------------------------------------------------
// k3: association + gated mean over cells, written straight into out[b,t,:]
// (which is also next step's ext). Wave loops all 16 cells -> no atomics.
// grid = 16 mtiles * 4 ngroups; 8 waves/block cover 8 ntiles.
// ---------------------------------------------------------------------------
__global__ __launch_bounds__(256) void k_assoc(
    const bf16_t* __restrict__ hnew,  // [C,B,H]
    const bf16_t* __restrict__ WaT,   // [C,H,H] (N,K)
    const float*  __restrict__ ba,    // [C,H]
    const float*  __restrict__ gate_logit, // [C]
    float* __restrict__ out, int t) { // out[b,t,:] slice
    int bx = blockIdx.x, mtile = bx >> 2, ng = bx & 3;
    int tid = threadIdx.x, lane = tid & 31, wave = tid >> 5;
    int ntile = ng * 8 + wave, j0 = ntile * 16;

    float y[8];
#pragma unroll
    for (int r = 0; r < 8; ++r) y[r] = 0.0f;

    for (int c = 0; c < Cc; ++c) {
        const bf16_t* A = hnew + ((size_t)c * Bb + mtile * 16) * Hh;
        const bf16_t* W = WaT + ((size_t)c * Hh + j0) * Hh;
        if (c + 1 < Cc)
            __builtin_prefetch(WaT + ((size_t)(c + 1) * Hh + j0) * Hh, 0, 1);
        v8f acc = {};
        v16bf bcur = load_frag_b(W, Hh, lane);
        for (int k = 0; k < 16; ++k) {
            v16bf a = load_frag_a(A + k * 32, Hh, lane);
            v16bf bnxt;
            if (k < 15) bnxt = load_frag_b(W + (k + 1) * 32, Hh, lane);
            acc = WMMA_BF16(a, bcur, acc);
            if (k < 15) bcur = bnxt;
        }
        float g = sigmoid_f(gate_logit[c]);
        float bj = ba[c * Hh + j0 + (lane & 15)];
#pragma unroll
        for (int r = 0; r < 8; ++r) y[r] += tanhf(acc[r] + bj) * g;
    }
    int j = j0 + (lane & 15);
#pragma unroll
    for (int r = 0; r < 8; ++r) {
        int b = mtile * 16 + ((lane >> 4) << 3) + r;
        out[((size_t)b * Tt + t) * Hh + j] = y[r] * (1.0f / Cc);
    }
}

// ---------------------------------------------------------------------------
// Host driver: prep (convert weights, embed GEMM, zero state), then
// 128 x (perception -> gates/LSTM -> association) on the stream.
// ---------------------------------------------------------------------------
extern "C" void kernel_launch(void* const* d_in, const int* in_sizes, int n_in,
                              void* d_out, int out_size, void* d_ws, size_t ws_size,
                              hipStream_t stream) {
    const int*   tokens  = (const int*)  d_in[0];
    const float* emb     = (const float*)d_in[1];
    const float* Wproj   = (const float*)d_in[2];
    const float* bproj   = (const float*)d_in[3];
    const float* Wp      = (const float*)d_in[4];
    const float* bp      = (const float*)d_in[5];
    const float* ln_g    = (const float*)d_in[6];
    const float* ln_b    = (const float*)d_in[7];
    const float* Wih     = (const float*)d_in[8];
    const float* bih     = (const float*)d_in[9];
    const float* Whh     = (const float*)d_in[10];
    const float* bhh     = (const float*)d_in[11];
    const float* Wa      = (const float*)d_in[12];
    const float* ba      = (const float*)d_in[13];
    const float* gatel   = (const float*)d_in[14];
    float* out = (float*)d_out;

    // workspace carve-up (256B aligned)
    char* w = (char*)d_ws;
    auto alloc = [&](size_t bytes) -> void* {
        void* p = (void*)w;
        w += (bytes + 255) & ~(size_t)255;
        return p;
    };
    float*  xs     = (float*) alloc((size_t)Tt * Bb * Hh * 4);   // 64 MB
    bf16_t* Pbf    = (bf16_t*)alloc((size_t)Cc * Bb * Hh * 2);
    bf16_t* hbf0   = (bf16_t*)alloc((size_t)Cc * Bb * Hh * 2);
    bf16_t* hbf1   = (bf16_t*)alloc((size_t)Cc * Bb * Hh * 2);
    float*  cst    = (float*) alloc((size_t)Cc * Bb * Hh * 4);
    float*  ext0   = (float*) alloc((size_t)Bb * Hh * 4);
    bf16_t* WpT    = (bf16_t*)alloc((size_t)Cc * Hh * Hh * 2);
    bf16_t* WaT    = (bf16_t*)alloc((size_t)Cc * Hh * Hh * 2);
    bf16_t* Wihb   = (bf16_t*)alloc((size_t)Cc * G4H * Hh * 2);
    bf16_t* Whhb   = (bf16_t*)alloc((size_t)Cc * G4H * Hh * 2);
    bf16_t* WprojT = (bf16_t*)alloc((size_t)Hh * Ee * 2);
    float*  bihh   = (float*) alloc((size_t)Cc * G4H * 4);

    auto grid1 = [](int n) { return dim3((n + 255) / 256); };

    // ---- prep: zero state, convert weights ----
    int nState = Cc * Bb * Hh;
    k_zero_f32<<<grid1(nState), 256, 0, stream>>>(cst, nState);
    k_zero_bf <<<grid1(nState), 256, 0, stream>>>(hbf0, nState);
    k_zero_f32<<<grid1(Bb * Hh), 256, 0, stream>>>(ext0, Bb * Hh);

    int nHH = Cc * Hh * Hh, nG = Cc * G4H * Hh;
    k_cvt_transpose_chh<<<grid1(nHH), 256, 0, stream>>>(Wp, WpT, nHH);
    k_cvt_transpose_chh<<<grid1(nHH), 256, 0, stream>>>(Wa, WaT, nHH);
    k_cvt_bf16<<<grid1(nG), 256, 0, stream>>>(Wih, Wihb, nG);
    k_cvt_bf16<<<grid1(nG), 256, 0, stream>>>(Whh, Whhb, nG);
    k_cvt_transpose_proj<<<grid1(Hh * Ee), 256, 0, stream>>>(Wproj, WprojT, Hh * Ee);
    k_bias_sum<<<grid1(Cc * G4H), 256, 0, stream>>>(bih, bhh, bihh, Cc * G4H);

    // ---- embedding + projection: xs[T,B,H] ----
    k_embed_proj<<<dim3((Tt * Bb) / 16), 256, 0, stream>>>(tokens, emb, WprojT, bproj, xs);

    // ---- recurrence over T ----
    for (int t = 0; t < Tt; ++t) {
        const float* ext = (t == 0) ? ext0 : (out + (size_t)(t - 1) * Hh);
        int ext_stride   = (t == 0) ? Hh : Tt * Hh;   // out is [B,T,H]
        const bf16_t* hp = (t & 1) ? hbf1 : hbf0;
        bf16_t*       hn = (t & 1) ? hbf0 : hbf1;

        k_perception<<<dim3(Cc * 16), 256, 0, stream>>>(
            xs + (size_t)t * Bb * Hh, ext, ext_stride, WpT, bp, ln_g, ln_b, Pbf);

        k_gates_lstm<<<dim3(Cc * 8 * 4), 256, 0, stream>>>(
            Pbf, hp, Wihb, Whhb, bihh, cst, hn);

        k_assoc<<<dim3(16 * 4), 256, 0, stream>>>(
            hn, WaT, ba, gatel, out, t);
    }
}
